// SlotAttention_1752346657288
// MI455X (gfx1250) — compile-verified, matrix operand order checked by reading, not run on hardware
//
#include <hip/hip_runtime.h>
#include <math.h>

#define Bsz   32
#define Nn    4096
#define DINc  512
#define DSc   512
#define Hh    8
#define Kslot 8
#define HD    64
#define MHc   1024
#define NITER 3
#define ROWS  (Bsz*Nn)   // 131072
#define EPSF  1e-8f

typedef __attribute__((ext_vector_type(16))) __bf16 v16bf;
typedef __attribute__((ext_vector_type(8)))  float  v8f;
typedef __attribute__((ext_vector_type(16))) unsigned short u16x16;
typedef __attribute__((ext_vector_type(8)))  unsigned short u16x8;

// Hardware f32->bf16 convert (lowers to v_cvt_*bf16_f32 on gfx1250).
static __device__ __forceinline__ unsigned short f2bf(float f) {
  return __builtin_bit_cast(unsigned short, (__bf16)f);
}
static __device__ __forceinline__ float bf2f(unsigned short h) {
  return __builtin_bit_cast(float, ((unsigned int)h) << 16);
}
static __device__ __forceinline__ v8f zero8() {
  v8f r;
#pragma unroll
  for (int i = 0; i < 8; ++i) r[i] = 0.f;
  return r;
}
static __device__ __forceinline__ u16x16 zero_frag() {
  u16x16 r;
#pragma unroll
  for (int i = 0; i < 16; ++i) r[i] = 0;
  return r;
}
// Load one 16-bit fragment for this lane: two contiguous runs of 8 halves
// (K = base..base+7 and base+16..base+23 per ISA A/B 16-bit layouts).
static __device__ __forceinline__ u16x16 load_frag(const unsigned short* p) {
  u16x8 lo = *(const u16x8*)p;
  u16x8 hi = *(const u16x8*)(p + 16);
  u16x16 r;
#pragma unroll
  for (int i = 0; i < 8; ++i) { r[i] = lo[i]; r[8 + i] = hi[i]; }
  return r;
}
static __device__ __forceinline__ v8f wmma_bf16(u16x16 a, u16x16 b, v8f c) {
  return __builtin_amdgcn_wmma_f32_16x16x32_bf16(
      false, __builtin_bit_cast(v16bf, a),
      false, __builtin_bit_cast(v16bf, b),
      (short)0, c, false, false);
}
static __device__ __forceinline__ float sigmoidf_(float x) {
  return 1.f / (1.f + __expf(-x));
}

// ---------------------------------------------------------------------------
// Prep: WT[c][d] = bf16 of (c<512 ? Wk[d][c] : Wv[d][c-512]); WqT fp32 transpose.
__global__ void prep_kernel(const float* __restrict__ Wk, const float* __restrict__ Wv,
                            const float* __restrict__ Wq,
                            unsigned short* __restrict__ WT, float* __restrict__ WqT) {
  long i = (long)blockIdx.x * 256 + threadIdx.x;       // 786432 total
  if (i < (long)1024 * 512) {
    int c = (int)(i >> 9), d = (int)(i & 511);
    float v = (c < 512) ? Wk[(long)d * 512 + c] : Wv[(long)d * 512 + (c - 512)];
    WT[(long)c * 512 + d] = f2bf(v);
  } else {
    long j = i - (long)1024 * 512;
    int c = (int)(j >> 9), d = (int)(j & 511);
    WqT[(long)c * 512 + d] = Wq[(long)d * 512 + c];
  }
}

__global__ void init_slots_kernel(const float* __restrict__ s0, float* __restrict__ slots) {
  int i = blockIdx.x * 256 + threadIdx.x;              // 131072
  slots[i] = s0[i & 511];
}

// ---------------------------------------------------------------------------
// LN stats per input row (one wave per row, wave32 shuffle reduction).
__global__ void ln_stats_kernel(const float* __restrict__ x,
                                float* __restrict__ mu, float* __restrict__ rstd) {
  int wave = threadIdx.x >> 5, lane = threadIdx.x & 31;
  long row = (long)blockIdx.x * 8 + wave;
  const float* xp = x + row * DINc;
  float s = 0.f, sq = 0.f;
#pragma unroll
  for (int i = 0; i < 16; ++i) { float v = xp[lane + 32 * i]; s += v; sq += v * v; }
#pragma unroll
  for (int off = 16; off > 0; off >>= 1) {
    s  += __shfl_xor(s,  off, 32);
    sq += __shfl_xor(sq, off, 32);
  }
  if (lane == 0) {
    float m = s * (1.f / DINc);
    float var = sq * (1.f / DINc) - m * m;
    mu[row] = m;
    rstd[row] = rsqrtf(var + 1e-5f);
  }
}

// ---------------------------------------------------------------------------
// KV projection GEMM with fused input-LN.
// One wave = one 16-row tile x 128 output columns (8 accumulators): every
// A fragment (load + LN + bf16 cvt) feeds 8 back-to-back WMMAs.
// Block = 8 waves covering all 1024 output columns of one row tile.
__global__ void kv_gemm_kernel(const float* __restrict__ x,
                               const float* __restrict__ mu, const float* __restrict__ rstd,
                               const float* __restrict__ g, const float* __restrict__ be,
                               const unsigned short* __restrict__ WT,
                               const float* __restrict__ bk, const float* __restrict__ bv,
                               unsigned short* __restrict__ kbuf,
                               unsigned short* __restrict__ vbuf) {
  __shared__ float tile[8][16][16];
  int wave = threadIdx.x >> 5, lane = threadIdx.x & 31;
  int rowTile = blockIdx.x;                            // 8192
  int rhalf = lane >> 4, l16 = lane & 15;

  long rowA = (long)rowTile * 16 + l16;
  float m = mu[rowA], rs = rstd[rowA];
  const float* xp = x + rowA * DINc;
  // wave handles columns [wave*128, wave*128+128)
  const unsigned short* wbase = WT + (long)(wave * 128 + l16) * 512;

  v8f acc[8];
#pragma unroll
  for (int j = 0; j < 8; ++j) acc[j] = zero8();

  for (int kk = 0; kk < 16; ++kk) {
    int d0 = kk * 32 + rhalf * 8;
    __builtin_prefetch(xp + d0 + 32, 0, 0);            // next k-step (global_prefetch_b8)
    // A fragment: fused LayerNorm + hardware bf16 cast
    v8f x0 = *(const v8f*)(xp + d0);
    v8f x1 = *(const v8f*)(xp + d0 + 16);
    v8f g0 = *(const v8f*)(g + d0);
    v8f g1 = *(const v8f*)(g + d0 + 16);
    v8f b0 = *(const v8f*)(be + d0);
    v8f b1 = *(const v8f*)(be + d0 + 16);
    u16x16 a;
#pragma unroll
    for (int i = 0; i < 8; ++i) {
      a[i]     = f2bf((x0[i] - m) * rs * g0[i] + b0[i]);
      a[8 + i] = f2bf((x1[i] - m) * rs * g1[i] + b1[i]);
    }
#pragma unroll
    for (int j = 0; j < 8; ++j) {
      u16x16 bfr = load_frag(wbase + (long)j * 16 * 512 + d0);
      acc[j] = wmma_bf16(a, bfr, acc[j]);
    }
  }

  // Writeback: 8 phases; per phase each wave parks one 16x16 f32 tile in LDS,
  // then 32 threads/tile repack + add bias and do one b128 bf16 store per thread.
  int t = threadIdx.x;
  int tw = t >> 5;
  int rrow = (t & 31) >> 1;
  int half = t & 1;
  long rowG = (long)rowTile * 16 + rrow;
#pragma unroll
  for (int j = 0; j < 8; ++j) {
#pragma unroll
    for (int v = 0; v < 8; ++v) tile[wave][v + 8 * rhalf][l16] = acc[j][v];
    __syncthreads();
    int colBase = tw * 128 + j * 16 + half * 8;
    int cb = colBase & 511;
    const float* bias = (colBase < 512) ? bk : bv;
    unsigned short* dst = ((colBase < 512) ? kbuf : vbuf) + rowG * 512 + cb;
    u16x8 o;
#pragma unroll
    for (int e = 0; e < 8; ++e) o[e] = f2bf(tile[tw][rrow][half * 8 + e] + bias[cb + e]);
    *(u16x8*)dst = o;
    __syncthreads();
  }
}

// ---------------------------------------------------------------------------
// q = LN_s(slots) @ Wq + bq  (fp32 accumulate, bf16 result)
__global__ void q_kernel(const float* __restrict__ slots,
                         const float* __restrict__ lnsg, const float* __restrict__ lnsb,
                         const float* __restrict__ WqT, const float* __restrict__ bq,
                         unsigned short* __restrict__ qbuf) {
  int row = blockIdx.x;                 // 256 = b*K + kslot
  int t = threadIdx.x;                  // 128
  __shared__ float sln[512];
  __shared__ float red[128];
  __shared__ float stats[2];
  const float* sp = slots + (long)row * 512;
  float vv[4];
#pragma unroll
  for (int i = 0; i < 4; ++i) vv[i] = sp[t + 128 * i];
  red[t] = vv[0] + vv[1] + vv[2] + vv[3];
  __syncthreads();
  for (int s = 64; s > 0; s >>= 1) { if (t < s) red[t] += red[t + s]; __syncthreads(); }
  if (t == 0) stats[0] = red[0] * (1.f / 512.f);
  __syncthreads();
  float mean = stats[0];
  float d0 = vv[0] - mean, d1 = vv[1] - mean, d2 = vv[2] - mean, d3 = vv[3] - mean;
  red[t] = d0 * d0 + d1 * d1 + d2 * d2 + d3 * d3;
  __syncthreads();
  for (int s = 64; s > 0; s >>= 1) { if (t < s) red[t] += red[t + s]; __syncthreads(); }
  if (t == 0) stats[1] = rsqrtf(red[0] * (1.f / 512.f) + 1e-5f);
  __syncthreads();
  float rs = stats[1];
#pragma unroll
  for (int i = 0; i < 4; ++i) {
    int j = t + 128 * i;
    sln[j] = (vv[i] - mean) * rs * lnsg[j] + lnsb[j];
  }
  __syncthreads();
#pragma unroll
  for (int i = 0; i < 4; ++i) {
    int c = t + 128 * i;
    float acc = bq[c];
    const float* wr = WqT + (long)c * 512;
    for (int j = 0; j < 512; ++j) acc += sln[j] * wr[j];
    qbuf[(long)row * 512 + c] = f2bf(acc);
  }
}

// ---------------------------------------------------------------------------
// logits = k . q * scale via WMMA, softmax over K=8 (+EPS), store attn fp32.
__global__ void attn_kernel(const unsigned short* __restrict__ kbuf,
                            const unsigned short* __restrict__ qbuf,
                            float* __restrict__ attn) {
  __shared__ float lt[8][16][16];
  int wave = threadIdx.x >> 5, lane = threadIdx.x & 31;
  int tileIdx = blockIdx.x * 8 + wave;  // 65536 = bh*256 + nTile
  int bh = tileIdx >> 8;
  int nT = tileIdx & 255;
  int b = bh >> 3, h = bh & 7;
  int rhalf = lane >> 4, l16 = lane & 15;

  // B fragments from q (columns 8..15 padded with zeros)
  u16x16 bf0, bf1;
  if (l16 < Kslot) {
    const unsigned short* qp = qbuf + (long)(b * Kslot + l16) * 512 + h * HD + rhalf * 8;
    bf0 = load_frag(qp);
    bf1 = load_frag(qp + 32);
  } else {
    bf0 = zero_frag();
    bf1 = zero_frag();
  }
  // A fragments from k
  int n = nT * 16 + l16;
  const unsigned short* kp = kbuf + (long)(b * Nn + n) * 512 + h * HD + rhalf * 8;
  v8f acc = zero8();
  acc = wmma_bf16(load_frag(kp), bf0, acc);
  acc = wmma_bf16(load_frag(kp + 32), bf1, acc);

#pragma unroll
  for (int v = 0; v < 8; ++v) lt[wave][v + 8 * rhalf][l16] = acc[v] * 0.125f;
  __syncthreads();

  if (lane < 16) {
    int r = lane;
    float a8[8];
    float mx = -1e30f;
#pragma unroll
    for (int k = 0; k < 8; ++k) { a8[k] = lt[wave][r][k]; mx = fmaxf(mx, a8[k]); }
    float s = 0.f;
#pragma unroll
    for (int k = 0; k < 8; ++k) { a8[k] = __expf(a8[k] - mx); s += a8[k]; }
    float inv = 1.f / s;
#pragma unroll
    for (int k = 0; k < 8; ++k) a8[k] = a8[k] * inv + EPSF;
    long nr = (long)nT * 16 + r;
    float* ap = attn + ((long)bh * Nn + nr) * 8;
    float4 lo = make_float4(a8[0], a8[1], a8[2], a8[3]);
    float4 hi = make_float4(a8[4], a8[5], a8[6], a8[7]);
    *(float4*)ap = lo;
    *(float4*)(ap + 4) = hi;
  }
}

// Deterministic column sums over N (no float atomics).
__global__ void colsum_kernel(const float* __restrict__ attn, float* __restrict__ colsum) {
  int bh = blockIdx.x;                  // 256
  int t = threadIdx.x;                  // 256
  __shared__ float red[256];
  float loc[8];
#pragma unroll
  for (int k = 0; k < 8; ++k) loc[k] = 0.f;
  const float* ap = attn + (long)bh * Nn * 8;
  for (int n = t; n < Nn; n += 256) {
    const float* r = ap + (long)n * 8;
#pragma unroll
    for (int k = 0; k < 8; ++k) loc[k] += r[k];
  }
  for (int k = 0; k < 8; ++k) {
    red[t] = loc[k];
    __syncthreads();
    for (int s = 128; s > 0; s >>= 1) { if (t < s) red[t] += red[t + s]; __syncthreads(); }
    if (t == 0) colsum[bh * 8 + k] = red[0];
    __syncthreads();
  }
}

// upd[b,k,h*64+d] = (sum_n attn[n,k] * v[n,d]) / colsum[k]
__global__ void upd_kernel(const float* __restrict__ attn,
                           const unsigned short* __restrict__ vbuf,
                           const float* __restrict__ colsum, float* __restrict__ updb) {
  int bh = blockIdx.x;                  // 256
  int b = bh >> 3, h = bh & 7;
  int t = threadIdx.x;                  // 512
  int ks = t >> 6, d = t & 63;
  const float* ap = attn + (long)bh * Nn * 8 + ks;
  const unsigned short* vp = vbuf + (long)b * Nn * 512 + h * HD + d;
  float acc = 0.f;
#pragma unroll 4
  for (int n = 0; n < Nn; ++n) acc += ap[(long)n * 8] * bf2f(vp[(long)n * 512]);
  updb[(long)(b * Kslot + ks) * 512 + h * HD + d] = acc / colsum[bh * 8 + ks];
}

// GRU cell (fp32): slots_mid = (1-z)*tanh(in + r*hn) + z*prev
__global__ void gru_kernel(const float* __restrict__ updb, const float* __restrict__ prev,
                           const float* __restrict__ W_ih, const float* __restrict__ b_ih,
                           const float* __restrict__ W_hh, const float* __restrict__ b_hh,
                           float* __restrict__ slots_mid) {
  int row = blockIdx.x;                 // 256
  int t = threadIdx.x;                  // 256
  __shared__ float u[512];
  __shared__ float p[512];
  u[t] = updb[(long)row * 512 + t];       u[t + 256] = updb[(long)row * 512 + t + 256];
  p[t] = prev[(long)row * 512 + t];       p[t + 256] = prev[(long)row * 512 + t + 256];
  __syncthreads();
#pragma unroll
  for (int rep = 0; rep < 2; ++rep) {
    int j = t + 256 * rep;
    float ir = b_ih[j], iz = b_ih[512 + j], inn = b_ih[1024 + j];
    float hr = b_hh[j], hz = b_hh[512 + j], hn = b_hh[1024 + j];
    for (int d = 0; d < 512; ++d) {
      float ud = u[d], pd = p[d];
      const float* wi = W_ih + (long)d * 1536 + j;
      const float* wh = W_hh + (long)d * 1536 + j;
      ir += ud * wi[0];   iz += ud * wi[512];   inn += ud * wi[1024];
      hr += pd * wh[0];   hz += pd * wh[512];   hn += pd * wh[1024];
    }
    float r = sigmoidf_(ir + hr);
    float z = sigmoidf_(iz + hz);
    float nn2 = tanhf(inn + r * hn);
    slots_mid[(long)row * 512 + j] = (1.f - z) * nn2 + z * p[j];
  }
}

// MLP stage 1: h = relu(LN_m(slots_mid) @ W1 + b1)
__global__ void mlp1_kernel(const float* __restrict__ slots_mid,
                            const float* __restrict__ lnmg, const float* __restrict__ lnmb,
                            const float* __restrict__ W1, const float* __restrict__ b1,
                            float* __restrict__ hbuf) {
  int row = blockIdx.x;                 // 256
  int t = threadIdx.x;                  // 256
  __shared__ float m[512];
  __shared__ float red[256];
  __shared__ float stats[2];
  float a = slots_mid[(long)row * 512 + t];
  float bb = slots_mid[(long)row * 512 + t + 256];
  red[t] = a + bb;
  __syncthreads();
  for (int s = 128; s > 0; s >>= 1) { if (t < s) red[t] += red[t + s]; __syncthreads(); }
  if (t == 0) stats[0] = red[0] * (1.f / 512.f);
  __syncthreads();
  float mean = stats[0];
  float da = a - mean, db = bb - mean;
  red[t] = da * da + db * db;
  __syncthreads();
  for (int s = 128; s > 0; s >>= 1) { if (t < s) red[t] += red[t + s]; __syncthreads(); }
  if (t == 0) stats[1] = rsqrtf(red[0] * (1.f / 512.f) + 1e-5f);
  __syncthreads();
  float rs = stats[1];
  m[t]       = da * rs * lnmg[t] + lnmb[t];
  m[t + 256] = db * rs * lnmg[t + 256] + lnmb[t + 256];
  __syncthreads();
#pragma unroll
  for (int i = 0; i < 4; ++i) {
    int c = t + 256 * i;
    float acc = b1[c];
    for (int j = 0; j < 512; ++j) acc += m[j] * W1[(long)j * MHc + c];
    hbuf[(long)row * MHc + c] = fmaxf(acc, 0.f);
  }
}

// MLP stage 2: slots = slots_mid + h @ W2 + b2
__global__ void mlp2_kernel(const float* __restrict__ hbuf,
                            const float* __restrict__ W2, const float* __restrict__ b2,
                            const float* __restrict__ slots_mid, float* __restrict__ slots) {
  int row = blockIdx.x;                 // 256
  int t = threadIdx.x;                  // 256
  __shared__ float hh[1024];
#pragma unroll
  for (int i = 0; i < 4; ++i) hh[t + 256 * i] = hbuf[(long)row * MHc + t + 256 * i];
  __syncthreads();
#pragma unroll
  for (int i = 0; i < 2; ++i) {
    int c = t + 256 * i;
    float acc = b2[c];
    for (int j = 0; j < 1024; ++j) acc += hh[j] * W2[(long)j * 512 + c];
    slots[(long)row * 512 + c] = slots_mid[(long)row * 512 + c] + acc;
  }
}

// attn_mean over heads -> first output region
__global__ void mean_kernel(const float* __restrict__ attn, float* __restrict__ out) {
  long i = (long)blockIdx.x * 256 + threadIdx.x;   // 1048576
  int k = (int)(i & 7);
  int n = (int)((i >> 3) & 4095);
  int b = (int)(i >> 15);
  float s = 0.f;
#pragma unroll
  for (int h = 0; h < 8; ++h) s += attn[(((long)(b * 8 + h) * Nn) + n) * 8 + k];
  out[i] = s * 0.125f;
}

__global__ void copy_slots_kernel(const float* __restrict__ slots, float* __restrict__ out) {
  int i = blockIdx.x * 256 + threadIdx.x;          // 131072
  out[i] = slots[i];
}

// ---------------------------------------------------------------------------
extern "C" void kernel_launch(void* const* d_in, const int* in_sizes, int n_in,
                              void* d_out, int out_size, void* d_ws, size_t ws_size,
                              hipStream_t stream) {
  (void)in_sizes; (void)n_in; (void)out_size; (void)ws_size;
  const float* x       = (const float*)d_in[0];
  const float* slots0  = (const float*)d_in[1];
  const float* ln_in_g = (const float*)d_in[2];
  const float* ln_in_b = (const float*)d_in[3];
  const float* ln_s_g  = (const float*)d_in[4];
  const float* ln_s_b  = (const float*)d_in[5];
  const float* ln_m_g  = (const float*)d_in[6];
  const float* ln_m_b  = (const float*)d_in[7];
  const float* Wq = (const float*)d_in[8];  const float* bq = (const float*)d_in[9];
  const float* Wk = (const float*)d_in[10]; const float* bk = (const float*)d_in[11];
  const float* Wv = (const float*)d_in[12]; const float* bv = (const float*)d_in[13];
  const float* W_ih = (const float*)d_in[14]; const float* b_ih = (const float*)d_in[15];
  const float* W_hh = (const float*)d_in[16]; const float* b_hh = (const float*)d_in[17];
  const float* W1 = (const float*)d_in[18]; const float* b1 = (const float*)d_in[19];
  const float* W2 = (const float*)d_in[20]; const float* b2 = (const float*)d_in[21];
  float* out = (float*)d_out;

  char* w = (char*)d_ws;
  auto alloc = [&](size_t n) { char* p = w; w += (n + 255) & ~(size_t)255; return p; };
  unsigned short* WT   = (unsigned short*)alloc((size_t)1024 * 512 * 2);
  float* WqT           = (float*)alloc((size_t)512 * 512 * 4);
  float* mu            = (float*)alloc((size_t)ROWS * 4);
  float* rstd          = (float*)alloc((size_t)ROWS * 4);
  unsigned short* kbuf = (unsigned short*)alloc((size_t)ROWS * 512 * 2);
  unsigned short* vbuf = (unsigned short*)alloc((size_t)ROWS * 512 * 2);
  unsigned short* qbuf = (unsigned short*)alloc((size_t)256 * 512 * 2);
  float* attn          = (float*)alloc((size_t)Bsz * Hh * Nn * Kslot * 4);
  float* colsum        = (float*)alloc((size_t)Bsz * Hh * Kslot * 4);
  float* slots         = (float*)alloc((size_t)Bsz * Kslot * DSc * 4);
  float* slots_mid     = (float*)alloc((size_t)Bsz * Kslot * DSc * 4);
  float* updb          = (float*)alloc((size_t)Bsz * Kslot * DSc * 4);
  float* hbuf          = (float*)alloc((size_t)256 * MHc * 4);

  prep_kernel<<<3072, 256, 0, stream>>>(Wk, Wv, Wq, WT, WqT);
  init_slots_kernel<<<512, 256, 0, stream>>>(slots0, slots);
  ln_stats_kernel<<<ROWS / 8, 256, 0, stream>>>(x, mu, rstd);
  kv_gemm_kernel<<<8192, 256, 0, stream>>>(x, mu, rstd, ln_in_g, ln_in_b,
                                           WT, bk, bv, kbuf, vbuf);
  for (int it = 0; it < NITER; ++it) {
    q_kernel<<<256, 128, 0, stream>>>(slots, ln_s_g, ln_s_b, WqT, bq, qbuf);
    attn_kernel<<<8192, 256, 0, stream>>>(kbuf, qbuf, attn);
    colsum_kernel<<<256, 256, 0, stream>>>(attn, colsum);
    upd_kernel<<<256, 512, 0, stream>>>(attn, vbuf, colsum, updb);
    gru_kernel<<<256, 256, 0, stream>>>(updb, slots, W_ih, b_ih, W_hh, b_hh, slots_mid);
    mlp1_kernel<<<256, 256, 0, stream>>>(slots_mid, ln_m_g, ln_m_b, W1, b1, hbuf);
    mlp2_kernel<<<256, 256, 0, stream>>>(hbuf, W2, b2, slots_mid, slots);
  }
  mean_kernel<<<4096, 256, 0, stream>>>(attn, out);
  copy_slots_kernel<<<512, 256, 0, stream>>>(slots, out + (size_t)Bsz * Nn * Kslot);
}
